// SimpleLSTMCell_77919296684536
// MI455X (gfx1250) — matrix operand order repaired and checked by reading, not compiled, not run on hardware
//
#include <hip/hip_runtime.h>
#include <hip/hip_bf16.h>
#include <math.h>

typedef __attribute__((ext_vector_type(16))) _Float16 v16h;
typedef __attribute__((ext_vector_type(8)))  _Float16 v8h;
typedef __attribute__((ext_vector_type(4)))  _Float16 v4h;
typedef __attribute__((ext_vector_type(8)))  float    v8f;
typedef __attribute__((ext_vector_type(4)))  float    v4f;
typedef __attribute__((ext_vector_type(4)))  unsigned int v4u;
typedef __attribute__((ext_vector_type(8)))  int      v8i;
typedef __attribute__((ext_vector_type(4)))  int      v4i;

#define BATCH 8192
#define HID   1024
#define DIN   2048   // INPUT + HIDDEN
#define BM    128    // block tile rows (batch)
#define BN    32     // block tile h-columns (per gate)
#define BK    64     // K chunk
#define AS    72     // LDS row stride in halfs (64 + 8 pad)

#if __has_builtin(__builtin_amdgcn_tensor_load_to_lds)
#define HAVE_TDM 1
#else
#define HAVE_TDM 0
#endif

__device__ __forceinline__ float sigmoidf_fast(float v) {
    return 1.0f / (1.0f + __expf(-v));
}

// ---------------------------------------------------------------------------
// Path A: one-time f32 -> f16 conversion into workspace
//   comb16 : [8192][2048]  (x | h_prev)
//   wall16 : [4096][2048]  (W_f ; W_i ; W_c ; W_o)
// ---------------------------------------------------------------------------
__global__ __launch_bounds__(256) void cvt_f16_kernel(
    const float* __restrict__ x, const float* __restrict__ h_prev,
    const float* __restrict__ Wf, const float* __restrict__ Wi,
    const float* __restrict__ Wc, const float* __restrict__ Wo,
    _Float16* __restrict__ comb16, _Float16* __restrict__ wall16)
{
    constexpr size_t T1 = (size_t)BATCH * DIN;          // combined elements
    const size_t tid4 = ((size_t)blockIdx.x * 256 + threadIdx.x) * 4;
    if (tid4 < T1) {
        const size_t m = tid4 >> 11;
        const int    c = (int)(tid4 & 2047);
        const float* src = (c < 1024) ? (x + m * 1024 + c)
                                      : (h_prev + m * 1024 + (c - 1024));
        v4f v = *(const v4f*)src;
        v4h o;
        #pragma unroll
        for (int j = 0; j < 4; ++j) o[j] = (_Float16)v[j];
        *(v4h*)(comb16 + tid4) = o;
    } else {
        const size_t idx = tid4 - T1;
        const size_t r = idx >> 11;
        const int    c = (int)(idx & 2047);
        const int    g = (int)(r >> 10);
        const size_t rr = r & 1023;
        const float* W = (g == 0) ? Wf : (g == 1) ? Wi : (g == 2) ? Wc : Wo;
        v4f v = *(const v4f*)(W + rr * (size_t)DIN + c);
        v4h o;
        #pragma unroll
        for (int j = 0; j < 4; ++j) o[j] = (_Float16)v[j];
        *(v4h*)(wall16 + idx) = o;
    }
}

#if HAVE_TDM
// ---------------------------------------------------------------------------
// TDM tile load: D# per CDNA5 ISA.
// data_size=2B, LDS padding: 4 dwords after every 32 dwords (64 f16 row ->
// 72-half stride, bank-conflict free).
// 6-arg builtin: (g0 v4u, g1 v8i, g2 v4i, g3 v4i, g4 v8i [unused], cpol i32)
// ---------------------------------------------------------------------------
__device__ __forceinline__ void tdm_load_tile(
    unsigned lds_addr, unsigned long long gaddr,
    unsigned tensor_d0, unsigned tensor_d1,
    unsigned tile_d0, unsigned tile_d1, unsigned tile_d2,
    unsigned long long stride0, unsigned long long stride1,
    unsigned tensor_d2)
{
    v4u g0;
    g0[0] = 1u;                                             // count=1
    g0[1] = lds_addr;                                       // LDS byte address
    g0[2] = (unsigned)(gaddr & 0xffffffffu);                // global_addr lo
    g0[3] = (unsigned)((gaddr >> 32) & 0x1ffffffu)          // global_addr hi
          | (2u << 30);                                     // type=2 (image)
    v8i g1;
    g1[0] = (int)((1u << 16)      // data_size = 2 bytes
                | (1u << 20)      // pad_enable
                | (4u << 22)      // pad_interval: 32 dwords
                | (3u << 25));    // pad_amount: 4 dwords (16B)
    g1[1] = (int)((tensor_d0 & 0xffffu) << 16);
    g1[2] = (int)((tensor_d0 >> 16) | ((tensor_d1 & 0xffffu) << 16));
    g1[3] = (int)((tensor_d1 >> 16) | (tile_d0 << 16));
    g1[4] = (int)(tile_d1 | (tile_d2 << 16));
    g1[5] = (int)(stride0 & 0xffffffffu);
    g1[6] = (int)((unsigned)((stride0 >> 32) & 0xffffu)
                | (unsigned)((stride1 & 0xffffu) << 16));
    g1[7] = (int)((unsigned)(stride1 >> 16));
    v4i g2;
    g2[0] = (int)tensor_d2; g2[1] = 0; g2[2] = 0; g2[3] = 0;
    v4i g3 = (v4i){0, 0, 0, 0};
    v8i g4 = (v8i){0, 0, 0, 0, 0, 0, 0, 0};   // unused fifth operand group
    __builtin_amdgcn_tensor_load_to_lds(g0, g1, g2, g3, g4, 0);
}

// ---------------------------------------------------------------------------
// Path A GEMM: TDM-staged, double-buffered LDS, fused LSTM epilogue.
// ---------------------------------------------------------------------------
__global__ __launch_bounds__(256) void lstm_wmma_tdm_kernel(
    const _Float16* __restrict__ comb16, const _Float16* __restrict__ wall16,
    const float* __restrict__ c_prev,
    const float* __restrict__ bf, const float* __restrict__ bi,
    const float* __restrict__ bc, const float* __restrict__ bo,
    float* __restrict__ hOut, float* __restrict__ cOut)
{
    constexpr int LDS_A = BM * AS;          // halfs
    constexpr int LDS_W = 4 * BN * AS;      // halfs
    __shared__ _Float16 sm[2][LDS_A + LDS_W];

    const int t     = threadIdx.x;
    const int lane  = t & 31;
    const int wave  = t >> 5;
    const int wm    = wave & 3;         // 4 waves along M
    const int wn    = wave >> 2;        // 2 waves along N
    const int half  = lane >> 4;
    const int l16   = lane & 15;
    const int mBase = blockIdx.x * BM;
    const int hBase = blockIdx.y * BN;

    const unsigned long long combA = (unsigned long long)(uintptr_t)comb16;
    const unsigned long long wallA = (unsigned long long)(uintptr_t)wall16;

    v8f acc[4][2];
    #pragma unroll
    for (int g = 0; g < 4; ++g)
        #pragma unroll
        for (int mt = 0; mt < 2; ++mt)
            acc[g][mt] = (v8f){0.f, 0.f, 0.f, 0.f, 0.f, 0.f, 0.f, 0.f};

    auto issueTDM = [&](int kc, int buf) {
        const int kBase = kc * BK;
        const unsigned ldsA = (unsigned)(uintptr_t)&sm[buf][0];
        const unsigned ldsW = (unsigned)(uintptr_t)&sm[buf][LDS_A];
        // A tile: 2D, tile 64(k) x 128(m) of combined [8192 x 2048]
        tdm_load_tile(ldsA,
                      combA + ((unsigned long long)mBase * DIN + kBase) * 2ull,
                      DIN, BATCH, BK, BM, 0,
                      DIN, 0ull, 0);
        // W tile: 3D, tile 64(k) x 32(h) x 4(gate) of wall [4][1024][2048]
        tdm_load_tile(ldsW,
                      wallA + ((unsigned long long)hBase * DIN + kBase) * 2ull,
                      DIN, HID, BK, BN, 4,
                      DIN, (unsigned long long)DIN * HID, 4);
    };

    auto compute = [&](int buf) {
        const _Float16* lA = &sm[buf][0];
        const _Float16* lW = &sm[buf][LDS_A];
        // Load ALL fragments for the whole 64-K chunk into distinct registers
        // first (24 independent ds_load_b128), then run 16 back-to-back WMMAs.
        v16h afrag[2][2];   // [kk][mt]
        v16h bfrag[2][4];   // [kk][g]
        #pragma unroll
        for (int kk = 0; kk < 2; ++kk) {
            const int ks = kk * 32;
            #pragma unroll
            for (int mt = 0; mt < 2; ++mt) {
                const int row = wm * 32 + mt * 16 + l16;
                const _Float16* pa = &lA[row * AS + ks];
                v8h a0 = *(const v8h*)(pa + 8 * half);
                v8h a1 = *(const v8h*)(pa + 16 + 8 * half);
                afrag[kk][mt] = __builtin_shufflevector(a0, a1,
                    0,1,2,3,4,5,6,7,8,9,10,11,12,13,14,15);
            }
            #pragma unroll
            for (int g = 0; g < 4; ++g) {
                const int row = wn * 16 + l16;
                const _Float16* pb = &lW[g * (BN * AS) + row * AS + ks + 8 * half];
                v8h b0 = *(const v8h*)(pb);
                v8h b1 = *(const v8h*)(pb + 16);
                bfrag[kk][g] = __builtin_shufflevector(b0, b1,
                    0,1,2,3,4,5,6,7,8,9,10,11,12,13,14,15);
            }
        }
        #pragma unroll
        for (int kk = 0; kk < 2; ++kk)
            #pragma unroll
            for (int g = 0; g < 4; ++g)
                #pragma unroll
                for (int mt = 0; mt < 2; ++mt)
                    acc[g][mt] = __builtin_amdgcn_wmma_f32_16x16x32_f16(
                        false, afrag[kk][mt], false, bfrag[kk][g],
                        (short)0, acc[g][mt], false, false);
    };

    constexpr int NCHUNK = DIN / BK;    // 32
    int buf = 0;
    if (wave == 0) {
        issueTDM(0, 0);
        __builtin_amdgcn_s_wait_tensorcnt((short)0);
    }
    __syncthreads();
    for (int kc = 0; kc < NCHUNK; ++kc) {
        if (kc + 1 < NCHUNK && wave == 0) issueTDM(kc + 1, buf ^ 1);
        compute(buf);
        if (wave == 0) __builtin_amdgcn_s_wait_tensorcnt((short)0);
        __syncthreads();
        buf ^= 1;
    }

    // ---- fused LSTM epilogue ----
    const int hCol = hBase + wn * 16 + l16;
    const float B0 = bf[hCol];
    const float B1 = bi[hCol];
    const float B2 = bc[hCol];
    const float B3 = bo[hCol];

    #pragma unroll
    for (int mt = 0; mt < 2; ++mt) {
        #pragma unroll
        for (int r = 0; r < 8; ++r) {
            const int m = mBase + wm * 32 + mt * 16 + r + 8 * half;
            const size_t off = (size_t)m * HID + hCol;
            const float fv = sigmoidf_fast(acc[0][mt][r] + B0);
            const float iv = sigmoidf_fast(acc[1][mt][r] + B1);
            const float cc = tanhf(acc[2][mt][r] + B2);
            const float ov = sigmoidf_fast(acc[3][mt][r] + B3);
            const float cp = c_prev[off];
            const float cn = fv * cp + iv * cc;
            hOut[off] = ov * tanhf(cn);
            cOut[off] = cn;
        }
    }
}
#endif // HAVE_TDM

// ---------------------------------------------------------------------------
// Path B (fallback, no workspace needed): round-1 kernel, register staging.
// ---------------------------------------------------------------------------
__global__ __launch_bounds__(256) void lstm_wmma_kernel(
    const float* __restrict__ x, const float* __restrict__ h_prev,
    const float* __restrict__ c_prev,
    const float* __restrict__ Wf, const float* __restrict__ bf,
    const float* __restrict__ Wi, const float* __restrict__ bi,
    const float* __restrict__ Wc, const float* __restrict__ bc,
    const float* __restrict__ Wo, const float* __restrict__ bo,
    float* __restrict__ hOut, float* __restrict__ cOut)
{
    __shared__ _Float16 lA[BM * AS];
    __shared__ _Float16 lB[4 * BN * AS];

    const int t     = threadIdx.x;
    const int lane  = t & 31;
    const int wave  = t >> 5;
    const int wm    = wave & 3;
    const int wn    = wave >> 2;
    const int half  = lane >> 4;
    const int l16   = lane & 15;
    const int mBase = blockIdx.x * BM;
    const int hBase = blockIdx.y * BN;
    const int sRow  = t >> 4;
    const int sCol  = (t & 15) * 4;

    v8f acc[4][2];
    #pragma unroll
    for (int g = 0; g < 4; ++g)
        #pragma unroll
        for (int mt = 0; mt < 2; ++mt)
            acc[g][mt] = (v8f){0.f, 0.f, 0.f, 0.f, 0.f, 0.f, 0.f, 0.f};

    v4f aReg[8];
    v4f bReg[8];

    auto stageLoad = [&](int kc) {
        const int kBase = kc * BK;
        const float* srcA = (kBase < 1024) ? x : h_prev;
        const int kcol = kBase & 1023;
        #pragma unroll
        for (int p = 0; p < 8; ++p) {
            const int m = p * 16 + sRow;
            aReg[p] = *(const v4f*)(srcA + (size_t)(mBase + m) * 1024 + kcol + sCol);
        }
        #pragma unroll
        for (int p = 0; p < 8; ++p) {
            const float* W = (p < 2) ? Wf : (p < 4) ? Wi : (p < 6) ? Wc : Wo;
            const int n = (p & 1) * 16 + sRow;
            bReg[p] = *(const v4f*)(W + (size_t)(hBase + n) * (size_t)DIN + kBase + sCol);
        }
    };

    auto stageStore = [&]() {
        #pragma unroll
        for (int p = 0; p < 8; ++p) {
            v4h hv;
            #pragma unroll
            for (int j = 0; j < 4; ++j) hv[j] = (_Float16)aReg[p][j];
            *(v4h*)(&lA[(p * 16 + sRow) * AS + sCol]) = hv;
        }
        #pragma unroll
        for (int p = 0; p < 8; ++p) {
            v4h hv;
            #pragma unroll
            for (int j = 0; j < 4; ++j) hv[j] = (_Float16)bReg[p][j];
            *(v4h*)(&lB[(p >> 1) * (BN * AS) + ((p & 1) * 16 + sRow) * AS + sCol]) = hv;
        }
    };

    auto compute = [&]() {
        #pragma unroll
        for (int ks = 0; ks < BK; ks += 32) {
            v16h afrag[2];
            #pragma unroll
            for (int mt = 0; mt < 2; ++mt) {
                const int row = wm * 32 + mt * 16 + l16;
                const _Float16* pa = &lA[row * AS + ks];
                v8h a0 = *(const v8h*)(pa + 8 * half);
                v8h a1 = *(const v8h*)(pa + 16 + 8 * half);
                afrag[mt] = __builtin_shufflevector(a0, a1,
                    0,1,2,3,4,5,6,7,8,9,10,11,12,13,14,15);
            }
            v16h bfrag[4];
            #pragma unroll
            for (int g = 0; g < 4; ++g) {
                const int row = wn * 16 + l16;
                const _Float16* pb = &lB[g * (BN * AS) + row * AS + ks + 8 * half];
                v8h b0 = *(const v8h*)(pb);
                v8h b1 = *(const v8h*)(pb + 16);
                bfrag[g] = __builtin_shufflevector(b0, b1,
                    0,1,2,3,4,5,6,7,8,9,10,11,12,13,14,15);
            }
            #pragma unroll
            for (int g = 0; g < 4; ++g)
                #pragma unroll
                for (int mt = 0; mt < 2; ++mt)
                    acc[g][mt] = __builtin_amdgcn_wmma_f32_16x16x32_f16(
                        false, afrag[mt], false, bfrag[g],
                        (short)0, acc[g][mt], false, false);
        }
    };

    constexpr int NCHUNK = DIN / BK;
    stageLoad(0);
    stageStore();
    __syncthreads();
    for (int kc = 0; kc < NCHUNK; ++kc) {
        if (kc + 1 < NCHUNK) stageLoad(kc + 1);
        compute();
        __syncthreads();
        if (kc + 1 < NCHUNK) {
            stageStore();
            __syncthreads();
        }
    }

    const int hCol = hBase + wn * 16 + l16;
    const float B0 = bf[hCol];
    const float B1 = bi[hCol];
    const float B2 = bc[hCol];
    const float B3 = bo[hCol];

    #pragma unroll
    for (int mt = 0; mt < 2; ++mt) {
        #pragma unroll
        for (int r = 0; r < 8; ++r) {
            const int m = mBase + wm * 32 + mt * 16 + r + 8 * half;
            const size_t off = (size_t)m * HID + hCol;
            const float fv = sigmoidf_fast(acc[0][mt][r] + B0);
            const float iv = sigmoidf_fast(acc[1][mt][r] + B1);
            const float cc = tanhf(acc[2][mt][r] + B2);
            const float ov = sigmoidf_fast(acc[3][mt][r] + B3);
            const float cp = c_prev[off];
            const float cn = fv * cp + iv * cc;
            hOut[off] = ov * tanhf(cn);
            cOut[off] = cn;
        }
    }
}

extern "C" void kernel_launch(void* const* d_in, const int* in_sizes, int n_in,
                              void* d_out, int out_size, void* d_ws, size_t ws_size,
                              hipStream_t stream) {
    const float* x      = (const float*)d_in[0];
    const float* h_prev = (const float*)d_in[1];
    const float* c_prev = (const float*)d_in[2];
    const float* Wf = (const float*)d_in[3];
    const float* bf = (const float*)d_in[4];
    const float* Wi = (const float*)d_in[5];
    const float* bi = (const float*)d_in[6];
    const float* Wc = (const float*)d_in[7];
    const float* bc = (const float*)d_in[8];
    const float* Wo = (const float*)d_in[9];
    const float* bo = (const float*)d_in[10];

    float* hOut = (float*)d_out;
    float* cOut = hOut + (size_t)BATCH * HID;

    constexpr size_t COMB_ELTS = (size_t)BATCH * DIN;     // 16M
    constexpr size_t WALL_ELTS = (size_t)4 * HID * DIN;   // 8M
    constexpr size_t WS_NEEDED = (COMB_ELTS + WALL_ELTS) * sizeof(_Float16);

#if HAVE_TDM
    if (ws_size >= WS_NEEDED) {
        _Float16* comb16 = (_Float16*)d_ws;
        _Float16* wall16 = comb16 + COMB_ELTS;
        const int cvtBlocks = (int)((COMB_ELTS + WALL_ELTS) / (4 * 256)); // 24576
        cvt_f16_kernel<<<cvtBlocks, 256, 0, stream>>>(
            x, h_prev, Wf, Wi, Wc, Wo, comb16, wall16);
        dim3 grid(BATCH / BM, HID / BN);
        lstm_wmma_tdm_kernel<<<grid, 256, 0, stream>>>(
            comb16, wall16, c_prev, bf, bi, bc, bo, hOut, cOut);
        return;
    }
#endif
    dim3 grid(BATCH / BM, HID / BN);
    lstm_wmma_kernel<<<grid, 256, 0, stream>>>(
        x, h_prev, c_prev, Wf, bf, Wi, bi, Wc, bc, Wo, bo, hOut, cOut);
}
